// PeakMemoryGNN_75144747810880
// MI455X (gfx1250) — matrix operand order repaired and checked by reading, not compiled, not run on hardware
//
#include <hip/hip_runtime.h>
#include <hip/hip_bf16.h>
#include <stdint.h>

#define HEADS 4
#define CH    64
#define HC    256
#define F_IN  128

typedef __attribute__((ext_vector_type(16))) __bf16 bf16x16;
typedef __attribute__((ext_vector_type(8)))  __bf16 bf16x8;
typedef __attribute__((ext_vector_type(8)))  float  f32x8;

union FragBF { bf16x16 v; struct { bf16x8 lo, hi; } p; };

__device__ __forceinline__ __bf16 f2bf(float f) {
  unsigned u = __float_as_uint(f);
  unsigned r = u + 0x7FFFu + ((u >> 16) & 1u);   // round-to-nearest-even
  unsigned short hs = (unsigned short)(r >> 16);
  __bf16 o; __builtin_memcpy(&o, &hs, sizeof(o));
  return o;
}

// order-preserving float <-> uint mapping for atomicMax-based segment max
__device__ __forceinline__ unsigned f2ord(float f) {
  unsigned u = __float_as_uint(f);
  return (u & 0x80000000u) ? ~u : (u | 0x80000000u);
}
__device__ __forceinline__ float ord2f(unsigned u) {
  unsigned v = (u & 0x80000000u) ? (u ^ 0x80000000u) : ~u;
  return __uint_as_float(v);
}

// ---------------- conversion kernels ----------------

// W [K,HC] row-major f32  ->  Wt [HC,K] (column-major of W) bf16
__global__ void k_convert_w(const float* __restrict__ W, __bf16* __restrict__ Wt,
                            int K, int total) {
  int idx = blockIdx.x * blockDim.x + threadIdx.x;
  if (idx >= total) return;
  int k = idx / HC, n = idx % HC;
  Wt[(size_t)n * K + k] = f2bf(W[idx]);
}

__global__ void k_cvt_f32_bf16(const float* __restrict__ in, __bf16* __restrict__ out,
                               int total) {
  int idx = blockIdx.x * blockDim.x + threadIdx.x;
  if (idx >= total) return;
  out[idx] = f2bf(in[idx]);
}

// ---------------- bf16 WMMA GEMM:  D[N,HC] = A[N,K] @ W[K,HC] (+bias) (+addend) (relu) ----
// A bf16 row-major, Wt bf16 = W transposed (so B fragment loads are contiguous).
// grid = (Nrows/16, 2), block = 256 (8 waves); wave w handles col tile blockIdx.y*8+w.
__global__ void k_gemm(const __bf16* __restrict__ A,
                       const __bf16* __restrict__ Wt,
                       const float* __restrict__ bias,
                       const float* __restrict__ addend,   // nullable
                       float* __restrict__ D,
                       int K, int relu) {
  const int lane = threadIdx.x & 31;
  const int wave = threadIdx.x >> 5;
  const int rowBase = blockIdx.x << 4;
  const int colBase = ((blockIdx.y << 3) + wave) << 4;
  const int m    = lane & 15;
  const int half = lane >> 4;

  // A fragment (contiguous k order used here, matches B): lane m=row; elems e -> K=half*16+e
  // => 16 contiguous elements = two 16B loads.
  const __bf16* arow = A  + (size_t)(rowBase + m) * K + half * 16;
  // B fragment (contiguous k order used here, matches A): lane m=col; elems e -> K=half*16+e
  // => 16 contiguous elements = two 16B loads from transposed weights.
  const __bf16* bcol = Wt + (size_t)(colBase + m) * K + half * 16;

  f32x8 acc = {};
  for (int k0 = 0; k0 < K; k0 += 32) {
    FragBF a, b;
    a.p.lo = *(const bf16x8*)(arow + k0);
    a.p.hi = *(const bf16x8*)(arow + k0 + 8);
    b.p.lo = *(const bf16x8*)(bcol + k0);
    b.p.hi = *(const bf16x8*)(bcol + k0 + 8);
    acc = __builtin_amdgcn_wmma_f32_16x16x32_bf16(false, a.v, false, b.v,
                                                  (short)0, acc, false, false);
  }

  // C/D layout: lanes 0-15 rows 0..7 in VGPR r, lanes 16-31 rows 8..15.
  const int col = colBase + m;
  const float bv = bias[col];
  const int row0 = rowBase + half * 8;
#pragma unroll
  for (int r = 0; r < 8; ++r) {
    size_t o = (size_t)(row0 + r) * HC + col;
    float val = acc[r] + bv;
    if (addend) val += addend[o];
    if (relu)   val = fmaxf(val, 0.0f);
    D[o] = val;
  }
}

// ---------------- edge attention ----------------

// one wave per edge; lane covers 8 channels; lanes 8h..8h+7 = head h
__global__ void k_score(const float* __restrict__ Q, const float* __restrict__ Kf,
                        const int* __restrict__ srcA, const int* __restrict__ dstA,
                        int E, float* __restrict__ score, unsigned* __restrict__ mbits) {
  int e    = blockIdx.x * 8 + (threadIdx.x >> 5);
  int lane = threadIdx.x & 31;
  if (e >= E) return;
  int s = srcA[e], d = dstA[e];
  const float4* qr = (const float4*)(Q  + (size_t)d * HC) + lane * 2;
  const float4* kr = (const float4*)(Kf + (size_t)s * HC) + lane * 2;
  float4 q0 = qr[0], q1 = qr[1];
  float4 k0 = kr[0], k1 = kr[1];
  float p = q0.x*k0.x + q0.y*k0.y + q0.z*k0.z + q0.w*k0.w
          + q1.x*k1.x + q1.y*k1.y + q1.z*k1.z + q1.w*k1.w;
  p += __shfl_xor(p, 1, 32);
  p += __shfl_xor(p, 2, 32);
  p += __shfl_xor(p, 4, 32);
  if ((lane & 7) == 0) {
    int h = lane >> 3;
    float sc = p * 0.125f;          // 1/sqrt(64)
    score[(size_t)e * HEADS + h] = sc;
    atomicMax(&mbits[(size_t)d * HEADS + h], f2ord(sc));
  }
}

__global__ void k_expsum(const float* __restrict__ score, const unsigned* __restrict__ mbits,
                         const int* __restrict__ dstA, int E,
                         float* __restrict__ aexp, float* __restrict__ denom) {
  int idx = blockIdx.x * blockDim.x + threadIdx.x;
  if (idx >= E * HEADS) return;
  int e = idx >> 2, h = idx & 3;
  int d = dstA[e];
  float m  = ord2f(mbits[(size_t)d * HEADS + h]);
  float av = __expf(score[idx] - m);
  aexp[idx] = av;
  atomicAdd(&denom[(size_t)d * HEADS + h], av);
}

__global__ void k_scatter(const float* __restrict__ V, const float* __restrict__ aexp,
                          const float* __restrict__ denom,
                          const int* __restrict__ srcA, const int* __restrict__ dstA,
                          int E, float* __restrict__ agg) {
  int e    = blockIdx.x * 8 + (threadIdx.x >> 5);
  int lane = threadIdx.x & 31;
  if (e >= E) return;
  int s = srcA[e], d = dstA[e];
  int h = lane >> 3;
  float w = aexp[(size_t)e * HEADS + h] /
            (denom[(size_t)d * HEADS + h] + 1e-16f);
  const float4* vr = (const float4*)(V + (size_t)s * HC) + lane * 2;
  float4 v0 = vr[0], v1 = vr[1];
  float* out = agg + (size_t)d * HC + lane * 8;
  atomicAdd(out + 0, v0.x * w); atomicAdd(out + 1, v0.y * w);
  atomicAdd(out + 2, v0.z * w); atomicAdd(out + 3, v0.w * w);
  atomicAdd(out + 4, v1.x * w); atomicAdd(out + 5, v1.y * w);
  atomicAdd(out + 6, v1.z * w); atomicAdd(out + 7, v1.w * w);
}

// ---------------- pooling + MLP head ----------------

__global__ void k_pool(const float* __restrict__ hx, const int* __restrict__ batchv,
                       float* __restrict__ g, float* __restrict__ cnt, int N) {
  int idx = blockIdx.x * blockDim.x + threadIdx.x;
  if (idx >= N * HC) return;
  int n = idx >> 8, j = idx & 255;
  int b = batchv[n];
  atomicAdd(&g[(size_t)b * HC + j], hx[idx]);
  if (j == 0) atomicAdd(&cnt[b], 1.0f);
}

__global__ void k_fc1(const float* __restrict__ g, const float* __restrict__ cnt,
                      const float* __restrict__ W1, const float* __restrict__ b1,
                      float* __restrict__ g1, int Bg) {
  int idx = blockIdx.x * blockDim.x + threadIdx.x;
  if (idx >= Bg * CH) return;
  int b = idx / CH, j = idx % CH;
  float inv = 1.0f / fmaxf(cnt[b], 1.0f);
  float s = 0.0f;
  for (int k = 0; k < HC; ++k) s += g[(size_t)b * HC + k] * W1[(size_t)k * CH + j];
  g1[idx] = fmaxf(s * inv + b1[j], 0.0f);
}

__global__ void k_fc2(const float* __restrict__ g1, const float* __restrict__ W2,
                      const float* __restrict__ b2, float* __restrict__ out, int Bg) {
  int b = threadIdx.x;
  if (b >= Bg) return;
  float s = 0.0f;
  for (int k = 0; k < CH; ++k) s += g1[(size_t)b * CH + k] * W2[k];
  out[b] = s + b2[0];
}

// ---------------- host launcher ----------------

extern "C" void kernel_launch(void* const* d_in, const int* in_sizes, int n_in,
                              void* d_out, int out_size, void* d_ws, size_t ws_size,
                              hipStream_t stream) {
  (void)n_in; (void)ws_size;
  const float* x      = (const float*)d_in[0];
  const int*   edge   = (const int*)d_in[1];
  const int*   batchv = (const int*)d_in[2];
  const int N  = in_sizes[0] / F_IN;     // 50000 (multiple of 16)
  const int E  = in_sizes[1] / 2;        // 800000
  const int Bg = out_size;               // 64
  const int* srcA = edge;
  const int* dstA = edge + E;

  // params: d_in[3 + L*8 + 2*m] = W, +1 = b, for m in {q,k,v,skip}
  const float* Wlin[3][4]; const float* blin[3][4];
  for (int L = 0; L < 3; ++L)
    for (int m = 0; m < 4; ++m) {
      Wlin[L][m] = (const float*)d_in[3 + L * 8 + 2 * m];
      blin[L][m] = (const float*)d_in[3 + L * 8 + 2 * m + 1];
    }
  const float* W1 = (const float*)d_in[27];
  const float* b1 = (const float*)d_in[28];
  const float* W2 = (const float*)d_in[29];
  const float* b2 = (const float*)d_in[30];

  // workspace carve-out
  char* base = (char*)d_ws;
  size_t off = 0;
  auto carve = [&](size_t bytes) -> void* {
    void* p = base + off;
    off = (off + bytes + 255) & ~(size_t)255;
    return p;
  };
  __bf16*   Abf    = (__bf16*)  carve((size_t)N * HC * sizeof(__bf16));
  float*    Qb     = (float*)   carve((size_t)N * HC * sizeof(float)); // also h output
  float*    Kb     = (float*)   carve((size_t)N * HC * sizeof(float));
  float*    Vb     = (float*)   carve((size_t)N * HC * sizeof(float));
  float*    Agg    = (float*)   carve((size_t)N * HC * sizeof(float));
  float*    scoreB = (float*)   carve((size_t)E * HEADS * sizeof(float));
  float*    aexpB  = (float*)   carve((size_t)E * HEADS * sizeof(float));
  unsigned* mbits  = (unsigned*)carve((size_t)N * HEADS * sizeof(unsigned));
  float*    denom  = (float*)   carve((size_t)N * HEADS * sizeof(float));
  __bf16*   wt[3][4];
  for (int L = 0; L < 3; ++L) {
    int KL = (L == 0) ? F_IN : HC;
    for (int m = 0; m < 4; ++m)
      wt[L][m] = (__bf16*)carve((size_t)KL * HC * sizeof(__bf16));
  }
  float* g   = (float*)carve((size_t)Bg * HC * sizeof(float));
  float* cnt = (float*)carve((size_t)Bg * sizeof(float));
  float* g1  = (float*)carve((size_t)Bg * CH * sizeof(float));

  // convert weights (transposed) + input activations to bf16
  for (int L = 0; L < 3; ++L) {
    int KL = (L == 0) ? F_IN : HC;
    int tot = KL * HC;
    for (int m = 0; m < 4; ++m)
      k_convert_w<<<(tot + 255) / 256, 256, 0, stream>>>(Wlin[L][m], wt[L][m], KL, tot);
  }
  k_cvt_f32_bf16<<<(N * F_IN + 255) / 256, 256, 0, stream>>>(x, Abf, N * F_IN);

  // three TransformerConv layers
  for (int L = 0; L < 3; ++L) {
    int KL = (L == 0) ? F_IN : HC;
    dim3 gg(N / 16, 2);
    k_gemm<<<gg, 256, 0, stream>>>(Abf, wt[L][0], blin[L][0], nullptr, Qb, KL, 0);
    k_gemm<<<gg, 256, 0, stream>>>(Abf, wt[L][1], blin[L][1], nullptr, Kb, KL, 0);
    k_gemm<<<gg, 256, 0, stream>>>(Abf, wt[L][2], blin[L][2], nullptr, Vb, KL, 0);

    hipMemsetAsync(mbits, 0, (size_t)N * HEADS * sizeof(unsigned), stream);
    hipMemsetAsync(denom, 0, (size_t)N * HEADS * sizeof(float), stream);
    hipMemsetAsync(Agg,   0, (size_t)N * HC * sizeof(float), stream);

    k_score  <<<(E + 7) / 8, 256, 0, stream>>>(Qb, Kb, srcA, dstA, E, scoreB, mbits);
    k_expsum <<<(E * HEADS + 255) / 256, 256, 0, stream>>>(scoreB, mbits, dstA, E, aexpB, denom);
    k_scatter<<<(E + 7) / 8, 256, 0, stream>>>(Vb, aexpB, denom, srcA, dstA, E, Agg);

    // skip GEMM fused with +agg and ReLU; writes h into Qb (safe: Qb not an input)
    k_gemm<<<gg, 256, 0, stream>>>(Abf, wt[L][3], blin[L][3], Agg, Qb, KL, 1);
    k_cvt_f32_bf16<<<(N * HC + 255) / 256, 256, 0, stream>>>(Qb, Abf, N * HC);
  }

  // global mean pool + MLP head
  hipMemsetAsync(g,   0, (size_t)Bg * HC * sizeof(float), stream);
  hipMemsetAsync(cnt, 0, (size_t)Bg * sizeof(float), stream);
  k_pool<<<(N * HC + 255) / 256, 256, 0, stream>>>(Qb, batchv, g, cnt, N);
  k_fc1<<<(Bg * CH + 255) / 256, 256, 0, stream>>>(g, cnt, W1, b1, g1, Bg);
  k_fc2<<<1, 64, 0, stream>>>(g1, W2, b2, (float*)d_out, Bg);
}
